// ChannelAttention_40656160424212
// MI455X (gfx1250) — compile-verified, matrix operand order checked by reading, not run on hardware
//
#include <hip/hip_runtime.h>

// ---------------------------------------------------------------------------
// ChannelAttention (non-local block), B=4, C=512, N=64*64=4096, Cq=64
// Flash-attention style fused kernel using CDNA5 bf16 WMMA (wave32).
// ---------------------------------------------------------------------------

typedef __bf16 bf16_t;
typedef __attribute__((ext_vector_type(16))) __bf16 v16bf;
typedef __attribute__((ext_vector_type(8)))  __bf16 v8bf;
typedef __attribute__((ext_vector_type(8)))  float  v8f;

#define NN 4096   // spatial positions
#define CC 512    // channels
#define CQ 64     // q/k channels
#define BK 32     // key block

// WMMA wrapper: D = A(16x32 bf16) * B(32x16 bf16) + C(f32)
__device__ __forceinline__ v8f wmma_bf16(v16bf a, v16bf b, v8f c) {
    return __builtin_amdgcn_wmma_f32_16x16x32_bf16(
        /*neg_a=*/false, a, /*neg_b=*/false, b,
        /*c_mod=*/(short)0, c, /*reuse_a=*/false, /*reuse_b=*/false);
}

// Per-lane fragment load for 16-bit A/B operands of 16x16x32 WMMA.
// Caller passes p = row_base + (lane>=16 ? 8 : 0); hardware layout wants
// elements {p[0..7], p[16..23]} in the 8 VGPRs (2 bf16 each).
__device__ __forceinline__ v16bf load_frag(const bf16_t* p) {
    v8bf lo = *reinterpret_cast<const v8bf*>(p);
    v8bf hi = *reinterpret_cast<const v8bf*>(p + 16);
    v16bf r;
#pragma unroll
    for (int i = 0; i < 8; ++i) { r[i] = lo[i]; r[i + 8] = hi[i]; }
    return r;
}

// ---------------------------------------------------------------------------
// Kernel 0a: concat + convert weights to bf16.  wb rows: [0,64)=wq,
// [64,128)=wk, [128,640)=wv ; row length 512.
// ---------------------------------------------------------------------------
__global__ __launch_bounds__(256) void cvt_w_kernel(
    const float* __restrict__ wq, const float* __restrict__ wk,
    const float* __restrict__ wv, bf16_t* __restrict__ wb) {
    int i = blockIdx.x * 256 + threadIdx.x;
    if (i >= 640 * 512) return;
    int row = i >> 9, col = i & 511;
    float v;
    if (row < 64)        v = wq[row * 512 + col];
    else if (row < 128)  v = wk[(row - 64) * 512 + col];
    else                 v = wv[(row - 128) * 512 + col];
    wb[i] = (bf16_t)v;
}

// ---------------------------------------------------------------------------
// Kernel 0b: transpose-convert x [B,C,N] f32 -> xt [B,N,C] bf16 (LDS tiled).
// ---------------------------------------------------------------------------
__global__ __launch_bounds__(256) void trx_kernel(
    const float* __restrict__ x, bf16_t* __restrict__ xt) {
    __shared__ float tile[32][33];
    const int b  = blockIdx.z;
    const int c0 = blockIdx.y * 32;
    const int n0 = blockIdx.x * 32;
    const int tx = threadIdx.x, ty = threadIdx.y;   // block (32,8)
    const float* xb = x + (size_t)b * CC * NN;
#pragma unroll
    for (int i = 0; i < 4; ++i)
        tile[ty + i * 8][tx] = xb[(size_t)(c0 + ty + i * 8) * NN + n0 + tx];
    __syncthreads();
    bf16_t* xo = xt + (size_t)b * NN * CC;
#pragma unroll
    for (int i = 0; i < 4; ++i)
        xo[(size_t)(n0 + ty + i * 8) * CC + c0 + tx] = (bf16_t)tile[tx][ty + i * 8];
}

// ---------------------------------------------------------------------------
// Kernel 1: fused QKV projection with WMMA.  One wave = one 16x16 out tile.
//   oy 0..3  -> Q rows,  write qt [B,N,64] bf16 (query-major, WMMA A/B layout)
//   oy 4..7  -> K rows,  write kt [B,N,64] bf16
//   oy 8..39 -> V rows,  write vb [B,C,N]  bf16 (channel-major, WMMA A layout)
// ---------------------------------------------------------------------------
__global__ __launch_bounds__(32) void proj_kernel(
    const bf16_t* __restrict__ wb,
    const float* __restrict__ bq, const float* __restrict__ bk,
    const float* __restrict__ bv,
    const bf16_t* __restrict__ xt,
    bf16_t* __restrict__ qt, bf16_t* __restrict__ kt, bf16_t* __restrict__ vb) {
    const int n0   = blockIdx.x * 16;
    const int oy   = blockIdx.y;            // 0..39
    const int b    = blockIdx.z;
    const int lane = threadIdx.x;
    const int lrow = lane & 15;
    const int lhi  = lane >> 4;

    const bf16_t* wrow = wb + (size_t)(oy * 16 + lrow) * 512 + lhi * 8;
    const bf16_t* xrow = xt + ((size_t)b * NN + n0 + lrow) * 512 + lhi * 8;

    v8f acc = {};
#pragma unroll 4
    for (int c0 = 0; c0 < 512; c0 += 32) {
        v16bf a = load_frag(wrow + c0);
        v16bf bf = load_frag(xrow + c0);
        acc = wmma_bf16(a, bf, acc);
    }

    const float* bias; int obase; bf16_t* dst; int is_v;
    if (oy < 4)      { bias = bq; obase = oy * 16;        dst = qt; is_v = 0; }
    else if (oy < 8) { bias = bk; obase = (oy - 4) * 16;  dst = kt; is_v = 0; }
    else             { bias = bv; obase = (oy - 8) * 16;  dst = vb; is_v = 1; }

#pragma unroll
    for (int r = 0; r < 8; ++r) acc[r] += bias[obase + r + lhi * 8];

    const int n = n0 + lrow;
    if (!is_v) {
        // rows o = obase + lhi*8 + r are consecutive -> one b128 store
        bf16_t* p = dst + ((size_t)b * NN + n) * CQ + obase + lhi * 8;
        v8bf v;
#pragma unroll
        for (int r = 0; r < 8; ++r) v[r] = (bf16_t)acc[r];
        *reinterpret_cast<v8bf*>(p) = v;
    } else {
#pragma unroll
        for (int r = 0; r < 8; ++r) {
            const int c = obase + r + lhi * 8;
            dst[((size_t)b * CC + c) * NN + n] = (bf16_t)acc[r];
        }
    }
}

// ---------------------------------------------------------------------------
// Kernel 2: fused flash attention.  Block = 4 waves, 16 queries.
// Wave w owns channels [w*128, w*128+128) of the output; each wave computes
// the (identical) 16x32 S tile + online softmax locally, transposes P through
// its private LDS slice, then does 8 PV WMMAs per key block.
// ---------------------------------------------------------------------------
__global__ __launch_bounds__(128) void attn_kernel(
    const bf16_t* __restrict__ qt, const bf16_t* __restrict__ kt,
    const bf16_t* __restrict__ vb, const float* __restrict__ x,
    const float* __restrict__ gamma, float* __restrict__ out) {
    const int b    = blockIdx.y;
    const int n0   = blockIdx.x * 16;
    const int tid  = threadIdx.x;
    const int wave = tid >> 5;
    const int lane = tid & 31;
    const int lrow = lane & 15;
    const int lhi  = lane >> 4;
    const int cbase = wave * 128;

    __shared__ bf16_t lds_p[4][16 * BK];   // per-wave P^T staging
    __shared__ float  lds_b[4][16];        // per-wave per-query broadcast

    // Q fragments for this query tile (A operand, kept in VGPRs for the loop)
    const bf16_t* qbase = qt + ((size_t)b * NN + n0 + lrow) * CQ + lhi * 8;
    const v16bf aq0 = load_frag(qbase);        // c  0..31
    const v16bf aq1 = load_frag(qbase + 32);   // c 32..63

    v8f O[8];
#pragma unroll
    for (int t = 0; t < 8; ++t) O[t] = (v8f){};
    float Mrow[8], Lrow[8];
#pragma unroll
    for (int r = 0; r < 8; ++r) { Mrow[r] = -3.0e38f; Lrow[r] = 0.0f; }

    for (int m0 = 0; m0 < NN; m0 += BK) {
        // ---- S = Q^T K over this key block (2 tiles of 16 keys) ----
        v8f S[2];
#pragma unroll
        for (int mt = 0; mt < 2; ++mt) {
            const bf16_t* kb =
                kt + ((size_t)b * NN + m0 + mt * 16 + lrow) * CQ + lhi * 8;
            v8f s = {};
            s = wmma_bf16(aq0, load_frag(kb), s);
            s = wmma_bf16(aq1, load_frag(kb + 32), s);
            S[mt] = s;
        }
        // prefetch next key block rows into cache
        if (m0 + BK < NN) {
            __builtin_prefetch(
                kt + ((size_t)b * NN + m0 + BK + lrow) * CQ, 0, 1);
        }

        // ---- block row-max (rows live across 16 lanes) ----
        float newM[8], alpha[8];
#pragma unroll
        for (int r = 0; r < 8; ++r) {
            float m = fmaxf(S[0][r], S[1][r]);
#pragma unroll
            for (int off = 8; off >= 1; off >>= 1)
                m = fmaxf(m, __shfl_xor(m, off, 16));
            newM[r]  = fmaxf(Mrow[r], m);
            alpha[r] = __expf(Mrow[r] - newM[r]);
        }

        // ---- P = exp(S - newM); stage transposed bf16 into LDS ----
#pragma unroll
        for (int mt = 0; mt < 2; ++mt) {
#pragma unroll
            for (int r = 0; r < 8; ++r) {
                float p = __expf(S[mt][r] - newM[r]);
                S[mt][r] = p;
                const int q = r + lhi * 8;                 // query row
                lds_p[wave][q * BK + mt * 16 + lrow] = (bf16_t)p;
            }
        }
        // ---- row sums, update running softmax state ----
#pragma unroll
        for (int r = 0; r < 8; ++r) {
            float s = S[0][r] + S[1][r];
#pragma unroll
            for (int off = 8; off >= 1; off >>= 1)
                s += __shfl_xor(s, off, 16);
            Lrow[r] = Lrow[r] * alpha[r] + s;
            Mrow[r] = newM[r];
        }
        // broadcast alpha indexed by query to all lanes (column layout of O)
        if (lrow == 0) {
#pragma unroll
            for (int r = 0; r < 8; ++r) lds_b[wave][lhi * 8 + r] = alpha[r];
        }
        asm volatile("s_wait_dscnt 0x0" ::: "memory");
        const float alpha_q = lds_b[wave][lrow];
#pragma unroll
        for (int t = 0; t < 8; ++t)
#pragma unroll
            for (int r = 0; r < 8; ++r) O[t][r] *= alpha_q;

        // ---- PV: O[c, q] += V[c, m] * P^T[m, q] ----
        const v16bf pb = load_frag(&lds_p[wave][lrow * BK + lhi * 8]);
#pragma unroll
        for (int t = 0; t < 8; ++t) {
            const bf16_t* vp =
                vb + ((size_t)b * CC + cbase + t * 16 + lrow) * NN + m0 + lhi * 8;
            O[t] = wmma_bf16(load_frag(vp), pb, O[t]);
        }
    }

    // ---- final normalize + residual ----
    if (lrow == 0) {
#pragma unroll
        for (int r = 0; r < 8; ++r) lds_b[wave][lhi * 8 + r] = Lrow[r];
    }
    asm volatile("s_wait_dscnt 0x0" ::: "memory");
    const float linv = 1.0f / lds_b[wave][lrow];
    const float g = gamma[0];
#pragma unroll
    for (int t = 0; t < 8; ++t) {
#pragma unroll
        for (int r = 0; r < 8; ++r) {
            const int c = cbase + t * 16 + r + lhi * 8;
            const size_t idx = ((size_t)b * CC + c) * NN + (n0 + lrow);
            out[idx] = g * (O[t][r] * linv) + x[idx];
        }
    }
}

// ---------------------------------------------------------------------------
// Host-side launcher
// ---------------------------------------------------------------------------
extern "C" void kernel_launch(void* const* d_in, const int* in_sizes, int n_in,
                              void* d_out, int out_size, void* d_ws, size_t ws_size,
                              hipStream_t stream) {
    const float* x     = (const float*)d_in[0];
    const float* wq    = (const float*)d_in[1];
    const float* bq    = (const float*)d_in[2];
    const float* wk    = (const float*)d_in[3];
    const float* bk    = (const float*)d_in[4];
    const float* wv    = (const float*)d_in[5];
    const float* bv    = (const float*)d_in[6];
    const float* gamma = (const float*)d_in[7];
    float* out = (float*)d_out;

    char* ws = (char*)d_ws;
    // workspace layout (bytes), all 256B aligned; total ~38.4 MB
    bf16_t* wb = (bf16_t*)(ws);                       // 640*512*2   = 655360
    bf16_t* xt = (bf16_t*)(ws + 655360);              // 4*4096*512*2 = 16777216
    bf16_t* qt = (bf16_t*)(ws + 17432576);            // 4*4096*64*2  = 2097152
    bf16_t* kt = (bf16_t*)(ws + 19529728);            // 4*4096*64*2  = 2097152
    bf16_t* vb = (bf16_t*)(ws + 21626880);            // 4*512*4096*2 = 16777216

    cvt_w_kernel<<<(640 * 512 + 255) / 256, 256, 0, stream>>>(wq, wk, wv, wb);
    trx_kernel<<<dim3(NN / 32, CC / 32, 4), dim3(32, 8), 0, stream>>>(x, xt);
    proj_kernel<<<dim3(NN / 16, 40, 4), 32, 0, stream>>>(wb, bq, bk, bv, xt,
                                                         qt, kt, vb);
    attn_kernel<<<dim3(NN / 16, 4), 128, 0, stream>>>(qt, kt, vb, x, gamma, out);
}
